// MOTMPNet_1726576857600
// MI455X (gfx1250) — matrix operand order repaired and checked by reading, not compiled, not run on hardware
//
#include <hip/hip_runtime.h>
#include <hip/hip_bf16.h>

typedef __attribute__((ext_vector_type(16))) _Float16 v16h;
typedef __attribute__((ext_vector_type(8)))  _Float16 v8h;
typedef __attribute__((ext_vector_type(8)))  float    v8f;

#define NN 50000
#define NE 1000000
#define MT_NODE 3125   /* NN/16 */
#define ET 62500       /* NE/16 */

// combined packed edge-step weights resident in LDS (halves)
#define OFF_E1   0        /* 3*5*512 = 7680 */
#define OFF_E2   7680     /* 3*1*512 = 1536 */
#define OFF_FO1  9216     /* 2*4*512 = 4096 */
#define OFF_FO2  13312    /* 2*2*512 = 2048 */
#define OFF_FI1  15360    /* 4096 */
#define OFF_FI2  19456    /* 2048 */
#define EW_TOTAL 21504

#if __has_builtin(__builtin_amdgcn_global_load_async_to_lds_b128)
#define HAVE_ASYNC_LDS 1
// builtin expects int4 pointers with explicit address spaces:
// (int4 addrspace(1)* src, int4 addrspace(3)* dst, Ii offset, Ii cpol)
typedef int v4i_vs __attribute__((vector_size(16)));
typedef __attribute__((address_space(1))) v4i_vs* as1_v4i;
typedef __attribute__((address_space(3))) v4i_vs* as3_v4i;
#endif

// ---------------- WMMA helpers (CDNA5 wave32, 16x16x32 f16 -> f32) ----------------

static __device__ __forceinline__ v8f wmma16(v16h a, v16h b, v8f c) {
  return __builtin_amdgcn_wmma_f32_16x16x32_f16(false, a, false, b, (short)0, c, false, false);
}

// Packed-B layout: [nt][kc][lane][16 halves]; element i holds W[kc*32 + (i<8 ? (lane>>4)*8 + i : 16 + (lane>>4)*8 + i-8)][nt*16 + (lane&15)]
static __device__ __forceinline__ v16h ldB(const _Float16* pb, int nt, int kc, int nk, int lane) {
  return *(const v16h*)(pb + ((size_t)(nt * nk + kc) * 32 + lane) * 16);
}

// A fragment (16-bit A 16x32 layout): lane m=lane&15 is the row; hi=lane>>4 selects K offset.
// elements 0..7 -> K = kc*32 + hi*8 + i ; elements 8..15 -> K = kc*32 + 16 + hi*8 + i
static __device__ __forceinline__ v16h ldA_f16(const _Float16* rowbase, int kc, int hi) {
  v8h lo = *(const v8h*)(rowbase + kc * 32 + hi * 8);
  v8h up = *(const v8h*)(rowbase + kc * 32 + 16 + hi * 8);
  v16h a;
#pragma unroll
  for (int i = 0; i < 8; ++i) { a[i] = lo[i]; a[i + 8] = up[i]; }
  return a;
}

static __device__ __forceinline__ v16h ldA_f32(const float* rowbase, int kc, int hi) {
  const float* p = rowbase + kc * 32 + hi * 8;
  v16h a;
#pragma unroll
  for (int i = 0; i < 8; ++i) { a[i] = (_Float16)p[i]; a[i + 8] = (_Float16)p[16 + i]; }
  return a;
}

static __device__ __forceinline__ v8f bset(float b) {
  v8f r;
#pragma unroll
  for (int i = 0; i < 8; ++i) r[i] = b;
  return r;
}

static __device__ __forceinline__ void ds_wait()   { asm volatile("s_wait_dscnt 0"   ::: "memory"); }
static __device__ __forceinline__ void load_wait() { asm volatile("s_wait_loadcnt 0" ::: "memory"); }
static __device__ __forceinline__ void async_wait() {
#if __has_builtin(__builtin_amdgcn_s_wait_asynccnt)
  __builtin_amdgcn_s_wait_asynccnt(0);
#else
  asm volatile("s_wait_asynccnt 0" ::: "memory");
#endif
}

// ---------------- weight pre-pack: f32 [K,N] row-major -> f16 B fragments ----------------

__global__ void pack_b_kernel(const float* __restrict__ W, _Float16* __restrict__ dst,
                              int K, int N, int nk, int nt) {
  int idx = blockIdx.x * blockDim.x + threadIdx.x;
  int total = nt * nk * 32;
  if (idx >= total) return;
  int lane = idx & 31;
  int kc = (idx >> 5) % nk;
  int t  = idx / (32 * nk);
  int n  = t * 16 + (lane & 15);
  int hi = lane >> 4;
  _Float16* o = dst + (size_t)idx * 16;
#pragma unroll
  for (int i = 0; i < 16; ++i) {
    int k = kc * 32 + (i < 8 ? hi * 8 + i : 16 + hi * 8 + (i - 8));
    float v = (k < K && n < N) ? W[(size_t)k * N + n] : 0.0f;
    o[i] = (_Float16)v;
  }
}

__global__ void zero_kernel(float* __restrict__ p, int n) {
  int i = blockIdx.x * blockDim.x + threadIdx.x;
  if (i < n) p[i] = 0.0f;
}

// ---------------- enc_edge: 6 -> 18 -> 18 -> 16, scalar per edge ----------------

__global__ __launch_bounds__(256) void enc_edge_kernel(
    const float* __restrict__ ea,
    const float* __restrict__ W1, const float* __restrict__ B1,
    const float* __restrict__ W2, const float* __restrict__ B2,
    const float* __restrict__ W3, const float* __restrict__ B3,
    _Float16* __restrict__ hedge, _Float16* __restrict__ initedge) {
  __shared__ float w1[6 * 18], b1[18], w2[18 * 18], b2[18], w3[18 * 16], b3[16];
  for (int i = threadIdx.x; i < 108; i += 256) w1[i] = W1[i];
  for (int i = threadIdx.x; i < 324; i += 256) w2[i] = W2[i];
  for (int i = threadIdx.x; i < 288; i += 256) w3[i] = W3[i];
  if (threadIdx.x < 18) { b1[threadIdx.x] = B1[threadIdx.x]; b2[threadIdx.x] = B2[threadIdx.x]; }
  if (threadIdx.x < 16) b3[threadIdx.x] = B3[threadIdx.x];
  __syncthreads();
  int e = blockIdx.x * 256 + threadIdx.x;
  if (e >= NE) return;
  float in[6];
#pragma unroll
  for (int j = 0; j < 6; ++j) in[j] = ea[(size_t)e * 6 + j];
  float h1[18];
#pragma unroll
  for (int j = 0; j < 18; ++j) {
    float s = b1[j];
#pragma unroll
    for (int k = 0; k < 6; ++k) s += in[k] * w1[k * 18 + j];
    h1[j] = fmaxf(s, 0.0f);
  }
  float h2[18];
  for (int j = 0; j < 18; ++j) {
    float s = b2[j];
    for (int k = 0; k < 18; ++k) s += h1[k] * w2[k * 18 + j];
    h2[j] = fmaxf(s, 0.0f);
  }
#pragma unroll
  for (int j = 0; j < 16; ++j) {
    float s = b3[j];
    for (int k = 0; k < 18; ++k) s += h2[k] * w3[k * 16 + j];
    _Float16 v = (_Float16)fmaxf(s, 0.0f);
    hedge[(size_t)e * 16 + j] = v;
    initedge[(size_t)e * 16 + j] = v;
  }
}

// ---------------- enc_node layer1: [50000,2048] @ [2048,128] -> H1 f16 ----------------

__global__ __launch_bounds__(256) void enc_node_l1(
    const float* __restrict__ x, const _Float16* __restrict__ pb,
    const float* __restrict__ bias, _Float16* __restrict__ H1) {
  int wave = (blockIdx.x * 256 + threadIdx.x) >> 5;
  int lane = threadIdx.x & 31;
  if (wave >= MT_NODE) return;
  int hi = lane >> 4, cl = lane & 15;
  const float* rowbase = x + (size_t)(wave * 16 + cl) * 2048;
  v8f acc[8];
#pragma unroll
  for (int nt = 0; nt < 8; ++nt) acc[nt] = bset(bias[nt * 16 + cl]);
  for (int kc = 0; kc < 64; ++kc) {
    v16h a = ldA_f32(rowbase, kc, hi);
#pragma unroll
    for (int nt = 0; nt < 8; ++nt) acc[nt] = wmma16(a, ldB(pb, nt, kc, 64, lane), acc[nt]);
  }
#pragma unroll
  for (int nt = 0; nt < 8; ++nt)
#pragma unroll
    for (int r = 0; r < 8; ++r)
      H1[(size_t)(wave * 16 + r + hi * 8) * 128 + nt * 16 + cl] = (_Float16)fmaxf(acc[nt][r], 0.0f);
}

// ---------------- enc_node layer2: [50000,128] @ [128,32] -> h_node f16 ----------------

__global__ __launch_bounds__(256) void enc_node_l2(
    const _Float16* __restrict__ H1, const _Float16* __restrict__ pb,
    const float* __restrict__ bias, _Float16* __restrict__ hnode) {
  int wave = (blockIdx.x * 256 + threadIdx.x) >> 5;
  int lane = threadIdx.x & 31;
  if (wave >= MT_NODE) return;
  int hi = lane >> 4, cl = lane & 15;
  const _Float16* rowbase = H1 + (size_t)(wave * 16 + cl) * 128;
  v8f acc[2];
#pragma unroll
  for (int nt = 0; nt < 2; ++nt) acc[nt] = bset(bias[nt * 16 + cl]);
#pragma unroll
  for (int kc = 0; kc < 4; ++kc) {
    v16h a = ldA_f16(rowbase, kc, hi);
#pragma unroll
    for (int nt = 0; nt < 2; ++nt) acc[nt] = wmma16(a, ldB(pb, nt, kc, 4, lane), acc[nt]);
  }
#pragma unroll
  for (int nt = 0; nt < 2; ++nt)
#pragma unroll
    for (int r = 0; r < 8; ++r)
      hnode[(size_t)(wave * 16 + r + hi * 8) * 32 + nt * 16 + cl] = (_Float16)fmaxf(acc[nt][r], 0.0f);
}

// ---------------- per-step fused edge kernel (4 waves / block) ----------------
// Stages all packed edge/flow weights into LDS (async global->LDS when available),
// then: edge_mlp (WMMA) + classifier + flow_out/flow_in MLPs (WMMA) + masked atomic scatter.

__global__ __launch_bounds__(128) void edge_step_kernel(
    const _Float16* __restrict__ hnode, _Float16* __restrict__ hedge,
    const _Float16* __restrict__ initedge,
    const int* __restrict__ erow, const int* __restrict__ ecol,
    const _Float16* __restrict__ pW,      // EW_TOTAL halves, packed E1|E2|Fo1|Fo2|Fi1|Fi2
    const float* __restrict__ bE1, const float* __restrict__ bE2,
    const float* __restrict__ bFo1, const float* __restrict__ bFo2,
    const float* __restrict__ bFi1, const float* __restrict__ bFi2,
    const float* __restrict__ cW1, const float* __restrict__ cB1,
    const float* __restrict__ cW2, const float* __restrict__ cB2,
    float* __restrict__ sum_out, float* __restrict__ cnt_out,
    float* __restrict__ sum_in,  float* __restrict__ cnt_in,
    float* __restrict__ out, int cls_idx) {
  __shared__ __align__(16) _Float16 sW[EW_TOTAL];          // 43008 B weights
  __shared__ __align__(16) _Float16 smem[4][16 * 96 + 16 * 16];
  __shared__ int sR[4][16], sC[4][16];

  // ---- stage weights into LDS (all threads participate; uniform barrier) ----
  for (int i = threadIdx.x; i < EW_TOTAL / 8; i += 128) {
#ifdef HAVE_ASYNC_LDS
    __builtin_amdgcn_global_load_async_to_lds_b128(
        (as1_v4i)(void*)(pW + (size_t)i * 8),
        (as3_v4i)(void*)(sW + (size_t)i * 8), 0, 0);
#else
    *(v8h*)(sW + (size_t)i * 8) = *(const v8h*)(pW + (size_t)i * 8);
#endif
  }
#ifdef HAVE_ASYNC_LDS
  async_wait();
#endif
  __syncthreads();

  const _Float16* pE1  = sW + OFF_E1;
  const _Float16* pE2  = sW + OFF_E2;
  const _Float16* pFo1 = sW + OFF_FO1;
  const _Float16* pFo2 = sW + OFF_FO2;
  const _Float16* pFi1 = sW + OFF_FI1;
  const _Float16* pFi2 = sW + OFF_FI2;

  int w = threadIdx.x >> 5, lane = threadIdx.x & 31;
  int etile = blockIdx.x * 4 + w;
  if (etile >= ET) return;
  int hi = lane >> 4, cl = lane & 15;
  int e0 = etile * 16;
  _Float16* T = smem[w];              // 16 x 96 staging (transpose through LDS)
  _Float16* H = smem[w] + 16 * 96;    // 16 x 16 new h_edge tile

  int e_m = e0 + cl;                  // this lane's A row edge
  int rn = erow[e_m], cn = ecol[e_m];
  if (hi == 0) { sR[w][cl] = rn; sC[w][cl] = cn; }

  // ---- edge_mlp layer 1: A = [h_node[row] | h_node[col] | init_edge | h_edge], K=96, N=80 ----
  v16h a0 = ldA_f16(hnode + (size_t)rn * 32, 0, hi);
  v16h a1 = ldA_f16(hnode + (size_t)cn * 32, 0, hi);
  v8h ia = *(const v8h*)(initedge + (size_t)e_m * 16 + hi * 8);
  v8h ha = *(const v8h*)(hedge + (size_t)e_m * 16 + hi * 8);
  v16h a2;
#pragma unroll
  for (int i = 0; i < 8; ++i) { a2[i] = ia[i]; a2[i + 8] = ha[i]; }

  v8f acc[5];
#pragma unroll
  for (int nt = 0; nt < 5; ++nt) acc[nt] = bset(bE1[nt * 16 + cl]);
#pragma unroll
  for (int nt = 0; nt < 5; ++nt) {
    acc[nt] = wmma16(a0, ldB(pE1, nt, 0, 3, lane), acc[nt]);
    acc[nt] = wmma16(a1, ldB(pE1, nt, 1, 3, lane), acc[nt]);
    acc[nt] = wmma16(a2, ldB(pE1, nt, 2, 3, lane), acc[nt]);
  }
  // ReLU -> LDS (16x96), zero K-pad cols 80..95
  for (int i = lane; i < 16 * 16; i += 32) T[(i >> 4) * 96 + 80 + (i & 15)] = (_Float16)0.0f;
#pragma unroll
  for (int nt = 0; nt < 5; ++nt)
#pragma unroll
    for (int r = 0; r < 8; ++r)
      T[(r + hi * 8) * 96 + nt * 16 + cl] = (_Float16)fmaxf(acc[nt][r], 0.0f);
  ds_wait();

  int rrv[8], ccv[8];
#pragma unroll
  for (int r = 0; r < 8; ++r) { rrv[r] = sR[w][r + hi * 8]; ccv[r] = sC[w][r + hi * 8]; }

  // ---- edge_mlp layer 2: K=80 (pad 96), N=16 ----
  v8f e2 = bset(bE2[cl]);
#pragma unroll
  for (int kc = 0; kc < 3; ++kc)
    e2 = wmma16(ldA_f16(T + cl * 96, kc, hi), ldB(pE2, 0, kc, 3, lane), e2);
#pragma unroll
  for (int r = 0; r < 8; ++r)
    H[(r + hi * 8) * 16 + cl] = (_Float16)fmaxf(e2[r], 0.0f);
  ds_wait();

  // write new h_edge to global (coalesced b128 per lane)
  {
    int r2 = lane >> 1, half = lane & 1;
    *(v8h*)(hedge + (size_t)(e0 + r2) * 16 + half * 8) = *(const v8h*)(H + r2 * 16 + half * 8);
  }

  // ---- classifier 16 -> 8 -> 1 (steps >= first_class_step) ----
  if (cls_idx >= 0 && hi == 0) {
    float hv[16];
#pragma unroll
    for (int k = 0; k < 16; ++k) hv[k] = (float)H[cl * 16 + k];
    float s1[8];
#pragma unroll
    for (int j = 0; j < 8; ++j) {
      float s = cB1[j];
#pragma unroll
      for (int k = 0; k < 16; ++k) s += hv[k] * cW1[k * 8 + j];
      s1[j] = fmaxf(s, 0.0f);
    }
    float o = cB2[0];
#pragma unroll
    for (int j = 0; j < 8; ++j) o += s1[j] * cW2[j];
    out[(size_t)cls_idx * NE + (e0 + cl)] = o;
  }

  // ---- flow MLPs: base = [h_node[col] | h_edge_new], K=48 pad 64 ----
  v16h ab1;
  {
    v8h hb = *(const v8h*)(H + cl * 16 + hi * 8);
#pragma unroll
    for (int i = 0; i < 8; ++i) { ab1[i] = hb[i]; ab1[i + 8] = (_Float16)0.0f; }
  }

  // flow_out: 48 -> 56 (pad 64) -> 32 ; scatter masked by row < col
  {
    v8f fa[4];
#pragma unroll
    for (int nt = 0; nt < 4; ++nt) {
      int colg = nt * 16 + cl;
      fa[nt] = bset(colg < 56 ? bFo1[colg] : 0.0f);
    }
#pragma unroll
    for (int nt = 0; nt < 4; ++nt) {
      fa[nt] = wmma16(a1, ldB(pFo1, nt, 0, 2, lane), fa[nt]);
      fa[nt] = wmma16(ab1, ldB(pFo1, nt, 1, 2, lane), fa[nt]);
    }
#pragma unroll
    for (int nt = 0; nt < 4; ++nt)
#pragma unroll
      for (int r = 0; r < 8; ++r)
        T[(r + hi * 8) * 64 + nt * 16 + cl] = (_Float16)fmaxf(fa[nt][r], 0.0f);
    ds_wait();
    v8f f2[2];
#pragma unroll
    for (int nt = 0; nt < 2; ++nt) f2[nt] = bset(bFo2[nt * 16 + cl]);
#pragma unroll
    for (int kc = 0; kc < 2; ++kc) {
      v16h a = ldA_f16(T + cl * 64, kc, hi);
#pragma unroll
      for (int nt = 0; nt < 2; ++nt) f2[nt] = wmma16(a, ldB(pFo2, nt, kc, 2, lane), f2[nt]);
    }
#pragma unroll
    for (int nt = 0; nt < 2; ++nt)
#pragma unroll
      for (int r = 0; r < 8; ++r)
        if (rrv[r] < ccv[r])
          atomicAdd(&sum_out[(size_t)rrv[r] * 32 + nt * 16 + cl], fmaxf(f2[nt][r], 0.0f));
  }

  if (hi == 0) {
    if (rn < cn) atomicAdd(&cnt_out[rn], 1.0f);
    if (rn > cn) atomicAdd(&cnt_in[rn], 1.0f);
  }

  // flow_in: same base, flow_in weights, masked by row > col
  {
    v8f fa[4];
#pragma unroll
    for (int nt = 0; nt < 4; ++nt) {
      int colg = nt * 16 + cl;
      fa[nt] = bset(colg < 56 ? bFi1[colg] : 0.0f);
    }
#pragma unroll
    for (int nt = 0; nt < 4; ++nt) {
      fa[nt] = wmma16(a1, ldB(pFi1, nt, 0, 2, lane), fa[nt]);
      fa[nt] = wmma16(ab1, ldB(pFi1, nt, 1, 2, lane), fa[nt]);
    }
#pragma unroll
    for (int nt = 0; nt < 4; ++nt)
#pragma unroll
      for (int r = 0; r < 8; ++r)
        T[(r + hi * 8) * 64 + nt * 16 + cl] = (_Float16)fmaxf(fa[nt][r], 0.0f);
    ds_wait();
    v8f f2[2];
#pragma unroll
    for (int nt = 0; nt < 2; ++nt) f2[nt] = bset(bFi2[nt * 16 + cl]);
#pragma unroll
    for (int kc = 0; kc < 2; ++kc) {
      v16h a = ldA_f16(T + cl * 64, kc, hi);
#pragma unroll
      for (int nt = 0; nt < 2; ++nt) f2[nt] = wmma16(a, ldB(pFi2, nt, kc, 2, lane), f2[nt]);
    }
#pragma unroll
    for (int nt = 0; nt < 2; ++nt)
#pragma unroll
      for (int r = 0; r < 8; ++r)
        if (rrv[r] > ccv[r])
          atomicAdd(&sum_in[(size_t)rrv[r] * 32 + nt * 16 + cl], fmaxf(f2[nt][r], 0.0f));
  }
}

// ---------------- per-step node kernel: seg-mean + node_mlp (WMMA) + accumulator reset ----------------

__global__ __launch_bounds__(256) void node_step_kernel(
    float* __restrict__ sum_in, float* __restrict__ cnt_in,
    float* __restrict__ sum_out, float* __restrict__ cnt_out,
    const _Float16* __restrict__ pN, const float* __restrict__ bN,
    _Float16* __restrict__ hnode) {
  int w = threadIdx.x >> 5, lane = threadIdx.x & 31;
  int mtile = blockIdx.x * 8 + w;
  if (mtile >= MT_NODE) return;
  int hi = lane >> 4, cl = lane & 15;
  int node = mtile * 16 + cl;
  float rin = 1.0f / fmaxf(cnt_in[node], 1.0f);
  float rout = 1.0f / fmaxf(cnt_out[node], 1.0f);
  const float* si = sum_in + (size_t)node * 32;
  const float* so = sum_out + (size_t)node * 32;
  v16h a0, a1;
#pragma unroll
  for (int i = 0; i < 8; ++i) {
    a0[i] = (_Float16)(si[hi * 8 + i] * rin);
    a0[i + 8] = (_Float16)(si[16 + hi * 8 + i] * rin);
    a1[i] = (_Float16)(so[hi * 8 + i] * rout);
    a1[i + 8] = (_Float16)(so[16 + hi * 8 + i] * rout);
  }
  v8f acc[2];
#pragma unroll
  for (int nt = 0; nt < 2; ++nt) {
    acc[nt] = bset(bN[nt * 16 + cl]);
    acc[nt] = wmma16(a0, ldB(pN, nt, 0, 2, lane), acc[nt]);
    acc[nt] = wmma16(a1, ldB(pN, nt, 1, 2, lane), acc[nt]);
  }
#pragma unroll
  for (int nt = 0; nt < 2; ++nt)
#pragma unroll
    for (int r = 0; r < 8; ++r)
      hnode[(size_t)(mtile * 16 + r + hi * 8) * 32 + nt * 16 + cl] = (_Float16)fmaxf(acc[nt][r], 0.0f);
  // re-zero this wave's accumulator rows for the next step (after all reads)
  load_wait();
  int n0 = mtile * 16;
  for (int i = lane; i < 16 * 32; i += 32) {
    sum_in[(size_t)n0 * 32 + i] = 0.0f;
    sum_out[(size_t)n0 * 32 + i] = 0.0f;
  }
  if (hi == 0) { cnt_in[n0 + cl] = 0.0f; cnt_out[n0 + cl] = 0.0f; }
}

// ---------------- host launch ----------------

extern "C" void kernel_launch(void* const* d_in, const int* in_sizes, int n_in,
                              void* d_out, int out_size, void* d_ws, size_t ws_size,
                              hipStream_t stream) {
  // Input order: x, edge_attr, params (alphabetical pytree: classifier, edge_mlp, enc_edge,
  // enc_node, flow_in, flow_out, node_mlp; each (W,b) per layer), edge_index.
  const float* x  = (const float*)d_in[0];
  const float* ea = (const float*)d_in[1];
  const float* cW1 = (const float*)d_in[2];  const float* cB1 = (const float*)d_in[3];
  const float* cW2 = (const float*)d_in[4];  const float* cB2 = (const float*)d_in[5];
  const float* eW1 = (const float*)d_in[6];  const float* eB1 = (const float*)d_in[7];
  const float* eW2 = (const float*)d_in[8];  const float* eB2 = (const float*)d_in[9];
  const float* gW1 = (const float*)d_in[10]; const float* gB1 = (const float*)d_in[11];
  const float* gW2 = (const float*)d_in[12]; const float* gB2 = (const float*)d_in[13];
  const float* gW3 = (const float*)d_in[14]; const float* gB3 = (const float*)d_in[15];
  const float* nW1 = (const float*)d_in[16]; const float* nB1 = (const float*)d_in[17];
  const float* nW2 = (const float*)d_in[18]; const float* nB2 = (const float*)d_in[19];
  const float* fiW1 = (const float*)d_in[20]; const float* fiB1 = (const float*)d_in[21];
  const float* fiW2 = (const float*)d_in[22]; const float* fiB2 = (const float*)d_in[23];
  const float* foW1 = (const float*)d_in[24]; const float* foB1 = (const float*)d_in[25];
  const float* foW2 = (const float*)d_in[26]; const float* foB2 = (const float*)d_in[27];
  const float* mW  = (const float*)d_in[28]; const float* mB  = (const float*)d_in[29];
  const int* eidx = (const int*)d_in[30];
  const int* erow = eidx;
  const int* ecol = eidx + NE;
  float* out = (float*)d_out;

  // workspace carve
  char* ws = (char*)d_ws;
  size_t off = 0;
  auto carve = [&](size_t bytes) {
    void* p = ws + off;
    off = (off + bytes + 255) & ~(size_t)255;
    return p;
  };
  _Float16* hnode    = (_Float16*)carve((size_t)NN * 32 * 2);
  _Float16* H1       = (_Float16*)carve((size_t)NN * 128 * 2);
  _Float16* hedge    = (_Float16*)carve((size_t)NE * 16 * 2);
  _Float16* initedge = (_Float16*)carve((size_t)NE * 16 * 2);
  float* accum   = (float*)carve(((size_t)NN * 32 * 2 + NN * 2) * 4);
  float* sum_out = accum;
  float* sum_in  = accum + (size_t)NN * 32;
  float* cnt_out = accum + (size_t)NN * 64;
  float* cnt_in  = accum + (size_t)NN * 64 + NN;
  _Float16* pEn1 = (_Float16*)carve((size_t)64 * 8 * 512 * 2);
  _Float16* pEn2 = (_Float16*)carve((size_t)4 * 2 * 512 * 2);
  _Float16* pEdgeW = (_Float16*)carve((size_t)EW_TOTAL * 2);   // contiguous: E1|E2|Fo1|Fo2|Fi1|Fi2
  _Float16* pE1  = pEdgeW + OFF_E1;
  _Float16* pE2  = pEdgeW + OFF_E2;
  _Float16* pFo1 = pEdgeW + OFF_FO1;
  _Float16* pFo2 = pEdgeW + OFF_FO2;
  _Float16* pFi1 = pEdgeW + OFF_FI1;
  _Float16* pFi2 = pEdgeW + OFF_FI2;
  _Float16* pNm  = (_Float16*)carve((size_t)2 * 2 * 512 * 2);

  auto pack = [&](const float* W, _Float16* dst, int K, int N, int nk, int nt) {
    int total = nt * nk * 32;
    pack_b_kernel<<<(total + 255) / 256, 256, 0, stream>>>(W, dst, K, N, nk, nt);
  };
  pack(nW1, pEn1, 2048, 128, 64, 8);
  pack(nW2, pEn2, 128, 32, 4, 2);
  pack(eW1, pE1, 96, 80, 3, 5);
  pack(eW2, pE2, 80, 16, 3, 1);
  pack(foW1, pFo1, 48, 56, 2, 4);
  pack(foW2, pFo2, 56, 32, 2, 2);
  pack(fiW1, pFi1, 48, 56, 2, 4);
  pack(fiW2, pFi2, 56, 32, 2, 2);
  pack(mW, pNm, 64, 32, 2, 2);

  int nacc = NN * 64 + NN * 2;
  zero_kernel<<<(nacc + 255) / 256, 256, 0, stream>>>(accum, nacc);

  enc_edge_kernel<<<(NE + 255) / 256, 256, 0, stream>>>(ea, gW1, gB1, gW2, gB2, gW3, gB3,
                                                        hedge, initedge);
  enc_node_l1<<<(MT_NODE + 7) / 8, 256, 0, stream>>>(x, pEn1, nB1, H1);
  enc_node_l2<<<(MT_NODE + 7) / 8, 256, 0, stream>>>(H1, pEn2, nB2, hnode);

  for (int step = 1; step <= 12; ++step) {
    int cls = step - 2;  // first_class_step = 2 -> outputs 0..10
    edge_step_kernel<<<ET / 4, 128, 0, stream>>>(
        hnode, hedge, initedge, erow, ecol,
        pEdgeW,
        eB1, eB2, foB1, foB2, fiB1, fiB2,
        cW1, cB1, cW2, cB2,
        sum_out, cnt_out, sum_in, cnt_in,
        out, cls);
    node_step_kernel<<<(MT_NODE + 7) / 8, 256, 0, stream>>>(
        sum_in, cnt_in, sum_out, cnt_out, pNm, mB, hnode);
  }
}
